// DiscreteVAE_21492016350016
// MI455X (gfx1250) — matrix operand order repaired and checked by reading, not compiled, not run on hardware
//
#include <hip/hip_runtime.h>
#include <cstddef>
#include <cstdint>

// ---------------------------------------------------------------------------
// DiscreteVAE forward, MI455X (gfx1250, wave32).
// All activations are position-major [P, C]; every GEMM runs through
// v_wmma_f32_16x16x32_bf16 with 32x32 per-wave tiles, branch-free b128 loads
// and a branch-free epilogue (vectorized bias, cndmask relu).
// ---------------------------------------------------------------------------

typedef __attribute__((ext_vector_type(16))) __bf16 v16bf;
typedef __attribute__((ext_vector_type(8)))  float  v8f;

#define EPSV 1e-5f

// ---------------- workspace layout (float elements) ------------------------
static constexpr size_t OFF_A   = 0;           // 33,554,432 (x3 / gfeat / y1)
static constexpr size_t OFF_B   = 33554432;    // 16,777,216 (x1,x4 / gconv / h1,h2)
static constexpr size_t OFF_C   = 50331648;    // 18,874,368 (x2 / featdec[65536,288])
static constexpr size_t OFF_G   = 69206016;    // 16,777,216 (soft)
static constexpr size_t OFF_H   = 85983232;    // 33,554,432 (c3out / y2)
static constexpr size_t OFF_F   = 119537664;   //  4,718,592 (fcat [2048,2304])
static constexpr size_t OFF_E   = 124256256;   //    524,288 (encT / x_dec)
static constexpr size_t OFF_E2  = 124780544;   //    524,288 (sampled [2048,256])
static constexpr size_t OFF_D   = 125304832;   //    524,288 (gmax / f_in)
static constexpr size_t OFF_X0  = 125829120;   //  2,097,152 (X0 padded [65536,32])
static constexpr size_t OFF_FP  = 127926272;   //  2,097,152 (fine_pre [65536,32])
static constexpr size_t OFF_IDX = 130023424;   //      8,192 ints (knn idx)
static constexpr size_t OFF_PAD = 130031616;   //    262,144 (padded weights)
// pad sub-offsets
static constexpr size_t PAD_C1W = OFF_PAD;            // [128,32]
static constexpr size_t PAD_F1W = OFF_PAD + 4096;     // [512,288]
static constexpr size_t PAD_F3W = OFF_PAD + 151552;   // [32,512]
static constexpr size_t PAD_F3B = OFF_PAD + 167936;   // [32]

// ---------------- d_out layout (floats) ------------------------------------
static constexpr size_t OUT_WC     = 0;        // whole_coarse (32,512,3)
static constexpr size_t OUT_WF     = 49152;    // whole_fine   (32,2048,3)
static constexpr size_t OUT_COARSE = 245760;   // coarse       [2048,96]
static constexpr size_t OUT_FINE   = 250368;   // fine         (32,64,32,3)
static constexpr size_t OUT_LOGITS = 268800;   // logits       [2048,8192]

// ---------------- input indices --------------------------------------------
enum {
  IN_NEIGH = 0, IN_CENTER = 1, IN_NOISE = 2,
  ENC = 3, DG1 = 15, DG2 = 32, DEC = 49, CB = 65
};

// ===========================================================================
// Fragment helpers (contiguous k = 16*half + s, wave32)
// ===========================================================================
__device__ __forceinline__ v16bf pack16(float4 u0, float4 u1, float4 u2, float4 u3) {
  v16bf a;
  a[0]  = (__bf16)u0.x; a[1]  = (__bf16)u0.y; a[2]  = (__bf16)u0.z; a[3]  = (__bf16)u0.w;
  a[4]  = (__bf16)u1.x; a[5]  = (__bf16)u1.y; a[6]  = (__bf16)u1.z; a[7]  = (__bf16)u1.w;
  a[8]  = (__bf16)u2.x; a[9]  = (__bf16)u2.y; a[10] = (__bf16)u2.z; a[11] = (__bf16)u2.w;
  a[12] = (__bf16)u3.x; a[13] = (__bf16)u3.y; a[14] = (__bf16)u3.z; a[15] = (__bf16)u3.w;
  return a;
}
// A fragment (16x32, 16-bit): lane half 0 holds K 0..15, half 1 holds K 16..31
__device__ __forceinline__ v16bf load_afrag(const float* row, int k0, int hi) {
  const float4* q0 = (const float4*)(row + k0 + hi * 16);
  return pack16(q0[0], q0[1], q0[2], q0[3]);
}
// B fragment (32x16, 16-bit): lane half 0 holds K 0..15, half 1 holds K 16..31
__device__ __forceinline__ v16bf load_bfrag(const float* row, int k0, int hi) {
  const float4* q = (const float4*)(row + k0 + hi * 16);
  return pack16(q[0], q[1], q[2], q[3]);
}
__device__ __forceinline__ v8f wmma_bf16(v16bf a, v16bf b, v8f c) {
  return __builtin_amdgcn_wmma_f32_16x16x32_bf16(false, a, false, b, (short)0, c,
                                                 false, false);
}

// ===========================================================================
// gemm_nt: C[n*ldc + m] = act( sum_k A[m,k]*B[n,k] + bias[m] )
// A: [M,K] row-major (weights). B: [N,K] row-major (position-major acts).
// 32x32 tile per wave; M,N%32==0, K%32==0 (pre-padded) -> branch-free loop.
// ===========================================================================
__global__ void gemm_nt_wmma(const float* __restrict__ A, const float* __restrict__ B,
                             const float* __restrict__ bias, float* __restrict__ C,
                             int K, int lda, int ldb, int ldc,
                             int biasMode, int act, int Mt, int Nt) {
  int wave = threadIdx.x >> 5;
  long long tile = (long long)blockIdx.x * 8 + wave;
  if (tile >= (long long)Mt * Nt) return;   // wave-uniform
  int tm = (int)(tile / Nt);
  int tn = (int)(tile % Nt);
  int lane = threadIdx.x & 31;
  int hi = lane >> 4;
  int r  = lane & 15;

  const float* a0p = A + (size_t)(tm * 32 + r) * lda;
  const float* a1p = A + (size_t)(tm * 32 + 16 + r) * lda;
  const float* b0p = B + (size_t)(tn * 32 + r) * ldb;
  const float* b1p = B + (size_t)(tn * 32 + 16 + r) * ldb;

  v8f acc00 = {}, acc01 = {}, acc10 = {}, acc11 = {};
  for (int k0 = 0; k0 < K; k0 += 32) {
    __builtin_prefetch(a0p + k0 + 64, 0, 0);
    __builtin_prefetch(b0p + k0 + 64, 0, 0);
    v16bf a0 = load_afrag(a0p, k0, hi);
    v16bf a1 = load_afrag(a1p, k0, hi);
    v16bf b0 = load_bfrag(b0p, k0, hi);
    v16bf b1 = load_bfrag(b1p, k0, hi);
    acc00 = wmma_bf16(a0, b0, acc00);
    acc01 = wmma_bf16(a0, b1, acc01);
    acc10 = wmma_bf16(a1, b0, acc10);
    acc11 = wmma_bf16(a1, b1, acc11);
  }

  // epilogue: lane owns (m = mbase + v, n = subtile_n + r); branch-free:
  // bias fetched as two b128 vectors under a single uniform branch.
#pragma unroll
  for (int i = 0; i < 2; ++i) {            // M subtile
    int mbase = tm * 32 + i * 16 + hi * 8;
    float bv[8] = {0.f, 0.f, 0.f, 0.f, 0.f, 0.f, 0.f, 0.f};
    if (biasMode) {
      const float4* bp = (const float4*)(bias + mbase);
      float4 u0 = bp[0], u1 = bp[1];
      bv[0] = u0.x; bv[1] = u0.y; bv[2] = u0.z; bv[3] = u0.w;
      bv[4] = u1.x; bv[5] = u1.y; bv[6] = u1.z; bv[7] = u1.w;
    }
#pragma unroll
    for (int j = 0; j < 2; ++j) {          // N subtile
      v8f acc = (i == 0) ? (j == 0 ? acc00 : acc01) : (j == 0 ? acc10 : acc11);
      int n = tn * 32 + j * 16 + r;
      float va[8];
#pragma unroll
      for (int v = 0; v < 8; ++v) {
        float x  = acc[v] + bv[v];
        float xr = fmaxf(x, 0.f);
        va[v] = act ? xr : x;              // uniform cndmask, no branch
      }
      float4* p = (float4*)(C + (size_t)n * ldc + mbase);
      p[0] = make_float4(va[0], va[1], va[2], va[3]);
      p[1] = make_float4(va[4], va[5], va[6], va[7]);
    }
  }
}

// ===========================================================================
// gemm_nn: C[m*ldc + n] = sum_k A[m,k]*B[k,n]   (soft @ codebook)
// A contiguous along K; B strided (scalar loads). M,N,K %32 == 0.
// ===========================================================================
__global__ void gemm_nn_wmma(const float* __restrict__ A, const float* __restrict__ B,
                             float* __restrict__ C, int K, int lda, int ldb, int ldc,
                             int Mt, int Nt) {
  int wave = threadIdx.x >> 5;
  long long tile = (long long)blockIdx.x * 8 + wave;
  if (tile >= (long long)Mt * Nt) return;
  int tm = (int)(tile / Nt);
  int tn = (int)(tile % Nt);
  int lane = threadIdx.x & 31;
  int hi = lane >> 4;
  int r  = lane & 15;

  const float* a0p = A + (size_t)(tm * 32 + r) * lda;
  const float* a1p = A + (size_t)(tm * 32 + 16 + r) * lda;
  int n0 = tn * 32 + r;
  int n1 = tn * 32 + 16 + r;

  v8f acc00 = {}, acc01 = {}, acc10 = {}, acc11 = {};
  for (int k0 = 0; k0 < K; k0 += 32) {
    v16bf a0 = load_afrag(a0p, k0, hi);
    v16bf a1 = load_afrag(a1p, k0, hi);
    v16bf b0, b1;
#pragma unroll
    for (int h = 0; h < 16; ++h) {
      int k = k0 + hi * 16 + h;
      b0[h] = (__bf16)B[(size_t)k * ldb + n0];
      b1[h] = (__bf16)B[(size_t)k * ldb + n1];
    }
    acc00 = wmma_bf16(a0, b0, acc00);
    acc01 = wmma_bf16(a0, b1, acc01);
    acc10 = wmma_bf16(a1, b0, acc10);
    acc11 = wmma_bf16(a1, b1, acc11);
  }
#pragma unroll
  for (int i = 0; i < 2; ++i)
#pragma unroll
    for (int j = 0; j < 2; ++j) {
      v8f acc = (i == 0) ? (j == 0 ? acc00 : acc01) : (j == 0 ? acc10 : acc11);
      int n = tn * 32 + j * 16 + r;
#pragma unroll
      for (int v = 0; v < 8; ++v) {
        int m = tm * 32 + i * 16 + hi * 8 + v;
        C[(size_t)m * ldc + n] = acc[v];
      }
    }
}

// ---------------- batch-norm (+relu); X [P,C], stats per channel ------------
__global__ void bn_act(float* __restrict__ X, const float* __restrict__ g,
                       const float* __restrict__ b, long P, int C, int act) {
  int c = blockIdx.x;
  __shared__ float s1[256], s2[256];
  float sum = 0.f, sq = 0.f;
  for (long p = threadIdx.x; p < P; p += 256) { float v = X[(size_t)p * C + c]; sum += v; sq += v * v; }
  s1[threadIdx.x] = sum; s2[threadIdx.x] = sq;
  __syncthreads();
  for (int s = 128; s > 0; s >>= 1) {
    if ((int)threadIdx.x < s) { s1[threadIdx.x] += s1[threadIdx.x + s]; s2[threadIdx.x] += s2[threadIdx.x + s]; }
    __syncthreads();
  }
  float mean = s1[0] / (float)P;
  float var  = s2[0] / (float)P - mean * mean;
  float scale = rsqrtf(var + EPSV) * g[c];
  float shift = b[c] - mean * scale;
  for (long p = threadIdx.x; p < P; p += 256) {
    float v = X[(size_t)p * C + c] * scale + shift;
    if (act) v = fmaxf(v, 0.f);
    X[(size_t)p * C + c] = v;
  }
}

// ---------------- group-norm + leaky(0.2); X [B*Pb, C], block=(bat,grp) -----
__global__ void gn_leaky(float* __restrict__ X, const float* __restrict__ g,
                         const float* __restrict__ b, int C, int Pb, int ngroups) {
  int grp = blockIdx.x % ngroups;
  int bat = blockIdx.x / ngroups;
  int cg  = C / ngroups;
  int c0  = grp * cg;
  long col0 = (long)bat * Pb;
  long cnt  = (long)cg * Pb;
  __shared__ float s1[256], s2[256];
  float sum = 0.f, sq = 0.f;
  for (long i = threadIdx.x; i < cnt; i += 256) {
    int p = (int)(i / cg); int cc = (int)(i % cg);
    float v = X[(size_t)(col0 + p) * C + c0 + cc];
    sum += v; sq += v * v;
  }
  s1[threadIdx.x] = sum; s2[threadIdx.x] = sq;
  __syncthreads();
  for (int s = 128; s > 0; s >>= 1) {
    if ((int)threadIdx.x < s) { s1[threadIdx.x] += s1[threadIdx.x + s]; s2[threadIdx.x] += s2[threadIdx.x + s]; }
    __syncthreads();
  }
  float mean = s1[0] / (float)cnt;
  float var  = s2[0] / (float)cnt - mean * mean;
  float inv  = rsqrtf(var + EPSV);
  for (long i = threadIdx.x; i < cnt; i += 256) {
    int p = (int)(i / cg); int cc = (int)(i % cg);
    size_t o = (size_t)(col0 + p) * C + c0 + cc;
    float v = (X[o] - mean) * inv * g[c0 + cc] + b[c0 + cc];
    X[o] = v > 0.f ? v : 0.2f * v;
  }
}

// ---------------- kNN (k=4) over 64 nodes per batch -------------------------
__global__ void knn_k4(const float* __restrict__ center, int* __restrict__ idx) {
  int t = blockIdx.x * 256 + threadIdx.x;
  if (t >= 32 * 64) return;
  int b = t >> 6, m = t & 63;
  const float* cb = center + (size_t)b * 64 * 3;
  float mx = cb[m * 3], my = cb[m * 3 + 1], mz = cb[m * 3 + 2];
  int chosen[4];
  for (int kk = 0; kk < 4; ++kk) {
    float bv = -3.4e38f; int bi = 0;
    for (int n = 0; n < 64; ++n) {
      bool used = false;
      for (int j = 0; j < kk; ++j) used |= (chosen[j] == n);
      if (used) continue;
      float dx = cb[n * 3] - mx, dy = cb[n * 3 + 1] - my, dz = cb[n * 3 + 2] - mz;
      float nd = -(dx * dx + dy * dy + dz * dz);
      if (nd > bv) { bv = nd; bi = n; }
    }
    chosen[kk] = bi;
    idx[t * 4 + kk] = bi;
  }
}

// ---------------- graph feature: gf [8192, 2C] from f [2048, ldf] -----------
__global__ void graph_feat(const float* __restrict__ f, const int* __restrict__ idx,
                           float* __restrict__ gf, int C, int ldf) {
  long i = (long)blockIdx.x * 256 + threadIdx.x;
  int C2 = 2 * C;
  if (i >= (long)C2 * 8192) return;
  int col = (int)(i / C2);           // position (pg*4 + kk)
  int c2  = (int)(i % C2);
  int pg = col >> 2, kk = col & 3;
  float v;
  if (c2 < C) {
    int nb = (pg & ~63) + idx[pg * 4 + kk];
    v = f[(size_t)nb * ldf + c2] - f[(size_t)pg * ldf + c2];
  } else {
    v = f[(size_t)pg * ldf + (c2 - C)];
  }
  gf[(size_t)col * C2 + c2] = v;
}

// ---------------- max over k=4 neighbors: [8192,C] -> fcat cols -------------
__global__ void maxpool4(const float* __restrict__ gin, float* __restrict__ fout,
                         int C, int ldout) {
  long i = (long)blockIdx.x * 256 + threadIdx.x;
  if (i >= 2048L * C) return;
  int pg = (int)(i / C), c = (int)(i % C);
  const float* p = gin + (size_t)pg * 4 * C + c;
  float v = fmaxf(fmaxf(p[0], p[C]), fmaxf(p[2 * C], p[3 * C]));
  fout[(size_t)pg * ldout + c] = v;
}

// ---------------- max over N=32 points: [65536,C] -> [2048,C] ---------------
__global__ void maxpool32(const float* __restrict__ xin, float* __restrict__ out, int C) {
  long i = (long)blockIdx.x * 256 + threadIdx.x;
  if (i >= 2048L * C) return;
  int pg = (int)(i / C), c = (int)(i % C);
  const float* p = xin + (size_t)pg * 32 * C + c;
  float v = p[0];
#pragma unroll
  for (int j = 1; j < 32; ++j) v = fmaxf(v, p[(size_t)j * C]);
  out[(size_t)pg * C + c] = v;
}

// ---------------- X0 padded: [65536, 32] ------------------------------------
__global__ void build_x0pad(const float* __restrict__ ng, float* __restrict__ X0) {
  long i = (long)blockIdx.x * 256 + threadIdx.x;
  if (i >= 65536L * 32) return;
  int p = (int)(i / 32), d = (int)(i % 32);
  X0[i] = (d < 3) ? ng[(size_t)p * 3 + d] : 0.f;
}

// ---------------- x3 [65536,512] = concat(gmax bcast, x2) -------------------
__global__ void build_x3(const float* __restrict__ x2, const float* __restrict__ gmax,
                         float* __restrict__ x3) {
  long i = (long)blockIdx.x * 256 + threadIdx.x;
  if (i >= 65536L * 512) return;
  int p = (int)(i / 512), c = (int)(i % 512);
  x3[i] = (c < 256) ? gmax[(size_t)(p >> 5) * 256 + c]
                    : x2[(size_t)p * 256 + (c - 256)];
}

// ---------------- zero-padded weight copy -----------------------------------
__global__ void pad2d(const float* __restrict__ src, float* __restrict__ dst,
                      int Ms, int Ks, int Md, int Kd) {
  long i = (long)blockIdx.x * 256 + threadIdx.x;
  if (i >= (long)Md * Kd) return;
  int m = (int)(i / Kd), k = (int)(i % Kd);
  dst[i] = (m < Ms && k < Ks) ? src[(size_t)m * Ks + k] : 0.f;
}

// ---------------- gumbel softmax over 8192 tokens per row -------------------
__global__ void gumbel_softmax(const float* __restrict__ logits,
                               const float* __restrict__ noise,
                               float* __restrict__ soft) {
  int p = blockIdx.x;
  const float* l = logits + (size_t)p * 8192;
  const float* n = noise  + (size_t)p * 8192;
  float* o = soft + (size_t)p * 8192;
  __shared__ float sh[256];
  float m = -3.4e38f;
  for (int t = threadIdx.x; t < 8192; t += 256) m = fmaxf(m, l[t] + n[t]);
  sh[threadIdx.x] = m; __syncthreads();
  for (int s = 128; s > 0; s >>= 1) {
    if ((int)threadIdx.x < s) sh[threadIdx.x] = fmaxf(sh[threadIdx.x], sh[threadIdx.x + s]);
    __syncthreads();
  }
  m = sh[0]; __syncthreads();
  float sum = 0.f;
  for (int t = threadIdx.x; t < 8192; t += 256) { float e = expf(l[t] + n[t] - m); o[t] = e; sum += e; }
  sh[threadIdx.x] = sum; __syncthreads();
  for (int s = 128; s > 0; s >>= 1) {
    if ((int)threadIdx.x < s) sh[threadIdx.x] += sh[threadIdx.x + s];
    __syncthreads();
  }
  float inv = 1.f / sh[0];
  for (int t = threadIdx.x; t < 8192; t += 256) o[t] *= inv;
}

// ---------------- decode fold-input [65536, 288] (cols 261.. zero) ----------
__global__ void build_featdec(const float* __restrict__ xdec /*[2048,256]*/,
                              const float* __restrict__ coarse /*[2048,96]*/,
                              float* __restrict__ fd) {
  long i = (long)blockIdx.x * 256 + threadIdx.x;
  if (i >= 65536L * 288) return;
  int p = (int)(i / 288), c = (int)(i % 288);
  int pg = p >> 5, j = p & 31;
  float v;
  if (c < 256)        v = xdec[(size_t)pg * 256 + c];
  else if (c == 256)  v = (j & 1) ? 0.05f : -0.05f;
  else if (c == 257)  v = ((j >> 1) & 1) ? 0.05f : -0.05f;
  else if (c < 261)   v = coarse[(size_t)pg * 96 + (j >> 2) * 3 + (c - 258)];
  else                v = 0.f;
  fd[i] = v;
}

// ---------------- whole_coarse = coarse + center ----------------------------
__global__ void fin_coarse(const float* __restrict__ coarse,
                           const float* __restrict__ center, float* __restrict__ wc) {
  int i = blockIdx.x * 256 + threadIdx.x;
  if (i >= 49152) return;
  int d = i % 3; int p = i / 3;          // p = (b*64+g)*8 + ii
  int pg = p >> 3;
  wc[i] = coarse[i] + center[(size_t)pg * 3 + d];
}

// ---------------- fine = f3 + point_feat ; whole_fine = fine + center -------
__global__ void fin_fine(const float* __restrict__ fine_pre /*[65536,32]*/,
                         const float* __restrict__ coarse   /*[2048,96]*/,
                         const float* __restrict__ center,
                         float* __restrict__ fine_out, float* __restrict__ wf) {
  long i = (long)blockIdx.x * 256 + threadIdx.x;
  if (i >= 65536L * 3) return;
  int d = (int)(i % 3); long p = i / 3;
  long pg = p >> 5; int j = (int)(p & 31);
  float pf = coarse[(size_t)pg * 96 + (j >> 2) * 3 + d];
  float fv = fine_pre[(size_t)p * 32 + d] + pf;
  fine_out[i] = fv;
  wf[i] = fv + center[(size_t)pg * 3 + d];
}

// ===========================================================================
// Host-side orchestration
// ===========================================================================
static inline void gemm_nt(const float* A, const float* B, const float* bias, float* C,
                           int M, int N, int K, int lda, int ldb, int ldc,
                           int biasMode, int act, hipStream_t st) {
  int Mt = M / 32, Nt = N / 32;
  long long tiles = (long long)Mt * Nt;
  int blocks = (int)((tiles + 7) / 8);
  gemm_nt_wmma<<<blocks, 256, 0, st>>>(A, B, bias, C, K, lda, ldb, ldc, biasMode, act, Mt, Nt);
}
static inline void gemm_nn(const float* A, const float* B, float* C,
                           int M, int N, int K, int lda, int ldb, int ldc, hipStream_t st) {
  int Mt = M / 32, Nt = N / 32;
  long long tiles = (long long)Mt * Nt;
  int blocks = (int)((tiles + 7) / 8);
  gemm_nn_wmma<<<blocks, 256, 0, st>>>(A, B, C, K, lda, ldb, ldc, Mt, Nt);
}
static inline const float* FPtr(void* p) { return (const float*)p; }
static inline int GB(long n) { return (int)((n + 255) / 256); }

// One DGCNN: X [2048, cin] -> l5dst [2048, cout] (group-normed, leaky)
static void run_dgcnn(void* const* d_in, int pb, int cin, int cout,
                      const float* X, const int* knn, float* ws,
                      float* l5dst, hipStream_t st) {
  float* f_in  = ws + OFF_D;   // [2048, 128]
  float* fcat  = ws + OFF_F;   // [2048, 2304]
  float* gfeat = ws + OFF_A;   // [8192, 2*Cp]
  float* gconv = ws + OFF_B;   // [8192, Co]

  gemm_nt(FPtr(d_in[pb + 0]), X, FPtr(d_in[pb + 1]), f_in,
          128, 2048, cin, cin, cin, 128, 1, 0, st);

  const int Cs[5]     = {128, 256, 512, 512, 1024};
  const int coloff[4] = {0, 256, 768, 1280};
  const float* fprev = f_in;
  int ldf = 128;
  for (int L = 0; L < 4; ++L) {
    int Cp = Cs[L], Co = Cs[L + 1];
    int wi = pb + 2 + L * 3;
    graph_feat<<<GB(2L * Cp * 8192), 256, 0, st>>>(fprev, knn, gfeat, Cp, ldf);
    gemm_nt(FPtr(d_in[wi]), gfeat, nullptr, gconv,
            Co, 8192, 2 * Cp, 2 * Cp, 2 * Cp, Co, 0, 0, st);
    gn_leaky<<<32 * 4, 256, 0, st>>>(gconv, FPtr(d_in[wi + 1]), FPtr(d_in[wi + 2]),
                                     Co, 256, 4);
    maxpool4<<<GB(2048L * Co), 256, 0, st>>>(gconv, fcat + coloff[L], Co, 2304);
    fprev = fcat + coloff[L];
    ldf = 2304;
  }
  gemm_nt(FPtr(d_in[pb + 14]), fcat, nullptr, l5dst,
          cout, 2048, 2304, 2304, 2304, cout, 0, 0, st);
  gn_leaky<<<32 * 4, 256, 0, st>>>(l5dst, FPtr(d_in[pb + 15]), FPtr(d_in[pb + 16]),
                                   cout, 64, 4);
}

extern "C" void kernel_launch(void* const* d_in, const int* in_sizes, int n_in,
                              void* d_out, int out_size, void* d_ws, size_t ws_size,
                              hipStream_t stream) {
  (void)in_sizes; (void)n_in; (void)out_size; (void)ws_size;
  float* ws  = (float*)d_ws;
  float* out = (float*)d_out;

  // ---- pad irregular weights to WMMA-friendly shapes ----
  pad2d<<<GB(128L * 32), 256, 0, stream>>>(FPtr(d_in[ENC + 0]), ws + PAD_C1W, 128, 3, 128, 32);
  pad2d<<<GB(512L * 288), 256, 0, stream>>>(FPtr(d_in[DEC + 6]), ws + PAD_F1W, 512, 261, 512, 288);
  pad2d<<<GB(32L * 512), 256, 0, stream>>>(FPtr(d_in[DEC + 14]), ws + PAD_F3W, 3, 512, 32, 512);
  pad2d<<<1, 256, 0, stream>>>(FPtr(d_in[DEC + 15]), ws + PAD_F3B, 3, 1, 32, 1);

  // -------------------- encode --------------------
  build_x0pad<<<GB(65536L * 32), 256, 0, stream>>>(FPtr(d_in[IN_NEIGH]), ws + OFF_X0);
  // c1 + bn1 + relu : x1 [65536,128]
  gemm_nt(ws + PAD_C1W, ws + OFF_X0, FPtr(d_in[ENC + 1]), ws + OFF_B,
          128, 65536, 32, 32, 32, 128, 1, 0, stream);
  bn_act<<<128, 256, 0, stream>>>(ws + OFF_B, FPtr(d_in[ENC + 2]), FPtr(d_in[ENC + 3]),
                                  65536L, 128, 1);
  // c2 : x2 [65536,256]
  gemm_nt(FPtr(d_in[ENC + 4]), ws + OFF_B, FPtr(d_in[ENC + 5]), ws + OFF_C,
          256, 65536, 128, 128, 128, 256, 1, 0, stream);
  // gmax + concat -> x3 [65536,512]
  maxpool32<<<GB(2048L * 256), 256, 0, stream>>>(ws + OFF_C, ws + OFF_D, 256);
  build_x3<<<GB(65536L * 512), 256, 0, stream>>>(ws + OFF_C, ws + OFF_D, ws + OFF_A);
  // c3 + bn2 + relu : c3out [65536,512]
  gemm_nt(FPtr(d_in[ENC + 6]), ws + OFF_A, FPtr(d_in[ENC + 7]), ws + OFF_H,
          512, 65536, 512, 512, 512, 512, 1, 0, stream);
  bn_act<<<512, 256, 0, stream>>>(ws + OFF_H, FPtr(d_in[ENC + 8]), FPtr(d_in[ENC + 9]),
                                  65536L, 512, 1);
  // c4 : x4 [65536,256]; max over 32 -> encT [2048,256]
  gemm_nt(FPtr(d_in[ENC + 10]), ws + OFF_H, FPtr(d_in[ENC + 11]), ws + OFF_B,
          256, 65536, 512, 512, 512, 256, 1, 0, stream);
  maxpool32<<<GB(2048L * 256), 256, 0, stream>>>(ws + OFF_B, ws + OFF_E, 256);

  // -------------------- kNN (shared) --------------------
  int* knn = (int*)(ws + OFF_IDX);
  knn_k4<<<8, 256, 0, stream>>>(FPtr(d_in[IN_CENTER]), knn);

  // -------------------- dgcnn1 -> logits [2048,8192] in d_out --------------
  run_dgcnn(d_in, DG1, 256, 8192, ws + OFF_E, knn, ws, out + OUT_LOGITS, stream);

  // -------------------- gumbel softmax + sampling --------------------------
  gumbel_softmax<<<2048, 256, 0, stream>>>(out + OUT_LOGITS, FPtr(d_in[IN_NOISE]), ws + OFF_G);
  // sampled [2048,256] = soft [2048,8192] x codebook [8192,256]
  gemm_nn(ws + OFF_G, FPtr(d_in[CB]), ws + OFF_E2, 2048, 256, 8192, 8192, 256, 256, stream);

  // -------------------- dgcnn2 -> x_dec [2048,256] -------------------------
  run_dgcnn(d_in, DG2, 256, 256, ws + OFF_E2, knn, ws, ws + OFF_E, stream);

  // -------------------- decode --------------------
  float* h1 = ws + OFF_B;
  float* h2 = ws + OFF_B + 2097152;
  gemm_nt(FPtr(d_in[DEC + 0]), ws + OFF_E, FPtr(d_in[DEC + 1]), h1,
          1024, 2048, 256, 256, 256, 1024, 1, 1, stream);
  gemm_nt(FPtr(d_in[DEC + 2]), h1, FPtr(d_in[DEC + 3]), h2,
          1024, 2048, 1024, 1024, 1024, 1024, 1, 1, stream);
  gemm_nt(FPtr(d_in[DEC + 4]), h2, FPtr(d_in[DEC + 5]), out + OUT_COARSE,
          96, 2048, 1024, 1024, 1024, 96, 1, 0, stream);
  fin_coarse<<<GB(49152), 256, 0, stream>>>(out + OUT_COARSE, FPtr(d_in[IN_CENTER]),
                                            out + OUT_WC);
  build_featdec<<<GB(65536L * 288), 256, 0, stream>>>(ws + OFF_E, out + OUT_COARSE,
                                                      ws + OFF_C);
  // f1 + bn + relu : y1 [65536,512]
  gemm_nt(ws + PAD_F1W, ws + OFF_C, FPtr(d_in[DEC + 7]), ws + OFF_A,
          512, 65536, 288, 288, 288, 512, 1, 0, stream);
  bn_act<<<512, 256, 0, stream>>>(ws + OFF_A, FPtr(d_in[DEC + 8]), FPtr(d_in[DEC + 9]),
                                  65536L, 512, 1);
  // f2 + bn + relu : y2 [65536,512]
  gemm_nt(FPtr(d_in[DEC + 10]), ws + OFF_A, FPtr(d_in[DEC + 11]), ws + OFF_H,
          512, 65536, 512, 512, 512, 512, 1, 0, stream);
  bn_act<<<512, 256, 0, stream>>>(ws + OFF_H, FPtr(d_in[DEC + 12]), FPtr(d_in[DEC + 13]),
                                  65536L, 512, 1);
  // f3 : fine_pre [65536,32] (rows 0..2 valid)
  gemm_nt(ws + PAD_F3W, ws + OFF_H, ws + PAD_F3B, ws + OFF_FP,
          32, 65536, 512, 512, 512, 32, 1, 0, stream);
  fin_fine<<<GB(65536L * 3), 256, 0, stream>>>(ws + OFF_FP, out + OUT_COARSE,
                                               FPtr(d_in[IN_CENTER]),
                                               out + OUT_FINE, out + OUT_WF);
}
